// AttentionFuseEmbeddings_55301998903793
// MI455X (gfx1250) — matrix operand order repaired and checked by reading, not compiled, not run on hardware
//
#include <hip/hip_runtime.h>
#include <hip/hip_bf16.h>

// ---------------------------------------------------------------------------
// Attention block for MI455X (gfx1250, wave32, WMMA).
//   x = concat(e1,e2,e3)            [B,S,E]   B=8 S=2048 E=1024
//   q/k/v = x @ W{q,k,v}^T + b      (bf16 WMMA, f32 accum)
//   scores = q @ k^T  (per batch)   (bf16 WMMA, f32 accum, f32 stored)
//   probs  = softmax(scores)        (async global->LDS stage, f32 math, bf16 out)
//   w      = probs @ v              (bf16 WMMA; v stored transposed)
//   out    = leaky_relu(w @ Wf^T + bf)  -> f32
// GEMMs: 32x64 wave tiles; fully-unrolled k-loop, 2-deep pipelined buffers,
// single base pointer per operand with immediate-offset b128 loads.
// ---------------------------------------------------------------------------

#define Bn 8
#define Sn 2048
#define En 1024
#define E1n 256
#define E2n 256
#define E3n 512
#define NEG_SLOPE 0.2f

typedef __attribute__((ext_vector_type(16))) __bf16 v16bf;
typedef __attribute__((ext_vector_type(8)))  float  v8f;

__device__ __forceinline__ v8f wmma_bf16(v16bf a, v16bf b, v8f c) {
  // (neg_a, A, neg_b, B, c_mod, C, reuse_a, reuse_b)
  return __builtin_amdgcn_wmma_f32_16x16x32_bf16(false, a, false, b,
                                                 (short)0, c, false, false);
}

union FragU { v16bf v; uint4 u[2]; };

// Two 16B loads at (base+off) and (base+off+step2) -> one 16x32 / 32x16 fragment.
__device__ __forceinline__ v16bf frag2(const char* p, int off, int step2) {
  FragU f;
  f.u[0] = *reinterpret_cast<const uint4*>(p + off);
  f.u[1] = *reinterpret_cast<const uint4*>(p + off + step2);
  return f.v;
}

// A fragments for rows m0 and m0+16 (per-lane base pa, row stride arow bytes).
__device__ __forceinline__ void load_a2(const char* pa, int kb, int arow, v16bf a[2]) {
  a[0] = frag2(pa, kb, 32);
  a[1] = frag2(pa + arow, kb, 32);
}

// B fragments for 4 column groups n0+16j (per-lane base pb, group stride bcol).
__device__ __forceinline__ void load_b4(const char* pb, int kb, int bcol, v16bf b[4]) {
#pragma unroll
  for (int j = 0; j < 4; ++j) b[j] = frag2(pb + (size_t)j * bcol, kb, 32);
}

__device__ __forceinline__ void mma8(v8f acc[8], const v16bf a[2], const v16bf b[4]) {
#pragma unroll
  for (int j = 0; j < 4; ++j) {
    acc[j]     = wmma_bf16(a[0], b[j], acc[j]);
    acc[4 + j] = wmma_bf16(a[1], b[j], acc[4 + j]);
  }
}

// Per-wave GEMM: 32x64 output tile, K step 32, 8 accumulators.
// Fragment layouts (wave32):
//  A 16x32: lanes 0-15 -> M=lane, K {0..7,16..23}; lanes 16-31 -> K {8..15,24..31}
//  B 32x16: memory holds B[n][k] contiguous in k; lanes 0-15 K {0..7,16..23}, 16-31 K {8..15,24..31}
template<int LDA, int LDB, int KLEN>
__device__ __forceinline__ void gemm_accum32(const __bf16* __restrict__ A,
                                             const __bf16* __restrict__ Bp,
                                             int m0, int n0, int lane, v8f acc[8]) {
  const int r = lane & 15, sel = lane >> 4;
  const char* pa = (const char*)(A + (size_t)(m0 + r) * LDA + sel * 8);
  const char* pb = (const char*)(Bp + (size_t)(n0 + r) * LDB + sel * 8);
  constexpr int AROW = 32 * LDA;        // bytes between row-halves (16 rows)
  constexpr int BCOL = 32 * LDB;        // bytes between 16-col groups
  constexpr int NI   = KLEN / 32;       // k-groups (even)

  __builtin_prefetch(pa, 0, 3);
  __builtin_prefetch(pb, 0, 3);

  v16bf a0[2], b0[4], a1[2], b1[4];
  load_a2(pa, 0, AROW, a0);
  load_b4(pb, 0, BCOL, b0);
#pragma unroll
  for (int it = 0; it + 2 < NI; it += 2) {
    load_a2(pa, (it + 1) * 64, AROW, a1);
    load_b4(pb, (it + 1) * 64, BCOL, b1);
    mma8(acc, a0, b0);
    load_a2(pa, (it + 2) * 64, AROW, a0);
    load_b4(pb, (it + 2) * 64, BCOL, b0);
    mma8(acc, a1, b1);
  }
  load_a2(pa, (NI - 1) * 64, AROW, a1);
  load_b4(pb, (NI - 1) * 64, BCOL, b1);
  mma8(acc, a0, b0);
  mma8(acc, a1, b1);
}

// ---------------------------------------------------------------------------
// 1) concat embeddings -> bf16 x  [B*S, E]
__global__ void pack_x_kernel(const float* __restrict__ e1, const float* __restrict__ e2,
                              const float* __restrict__ e3, __bf16* __restrict__ xbf) {
  size_t total = (size_t)Bn * Sn * En;
  for (size_t idx = (size_t)blockIdx.x * blockDim.x + threadIdx.x; idx < total;
       idx += (size_t)gridDim.x * blockDim.x) {
    size_t row = idx / En;
    int c = (int)(idx % En);
    float v;
    if (c < E1n)            v = e1[row * E1n + c];
    else if (c < E1n + E2n) v = e2[row * E2n + (c - E1n)];
    else                    v = e3[row * E3n + (c - E1n - E2n)];
    xbf[idx] = (__bf16)v;
  }
}

// 2) convert Wq,Wk,Wv,Wf (f32 [out,in]) -> bf16, packed [4][E][E]
__global__ void pack_w_kernel(const float* __restrict__ wq, const float* __restrict__ wk,
                              const float* __restrict__ wv, const float* __restrict__ wf,
                              __bf16* __restrict__ wbf) {
  size_t total = (size_t)4 * En * En;
  for (size_t idx = (size_t)blockIdx.x * blockDim.x + threadIdx.x; idx < total;
       idx += (size_t)gridDim.x * blockDim.x) {
    int mat = (int)(idx / ((size_t)En * En));
    size_t off = idx % ((size_t)En * En);
    float v;
    if (mat == 0)      v = wq[off];
    else if (mat == 1) v = wk[off];
    else if (mat == 2) v = wv[off];
    else               v = wf[off];
    wbf[idx] = (__bf16)v;
  }
}

// 3) QKV projection GEMM. q,k row-major bf16 [B*S,E]; v stored transposed [b][e][s].
__global__ void qkv_gemm_kernel(const __bf16* __restrict__ x, const __bf16* __restrict__ w,
                                const float* __restrict__ bq, const float* __restrict__ bk,
                                const float* __restrict__ bv,
                                __bf16* __restrict__ q, __bf16* __restrict__ k,
                                __bf16* __restrict__ vT) {
  const int MT = (Bn * Sn) / 32;   // 512
  const int NT = En / 64;          // 16
  int wid  = blockIdx.x * (blockDim.x >> 5) + (threadIdx.x >> 5);
  int lane = threadIdx.x & 31;
  int mat = wid / (MT * NT);
  int rem = wid % (MT * NT);
  int m0 = (rem / NT) * 32;
  int n0 = (rem % NT) * 64;

  const __bf16* W = w + (size_t)mat * En * En;
  v8f acc[8] = {};
  gemm_accum32<En, En, En>(x, W, m0, n0, lane, acc);

  const float* bias = (mat == 0) ? bq : (mat == 1) ? bk : bv;
  int sel = lane >> 4, c = lane & 15;
#pragma unroll
  for (int h = 0; h < 2; ++h) {
#pragma unroll
    for (int j = 0; j < 4; ++j) {
      int col = n0 + 16 * j + c;
      float bb = bias[col];
      v8f av = acc[4 * h + j];
      if (mat == 2) {
        // vT[b][e][s]: 8 consecutive s per lane -> one 16B packed store
        int m = m0 + 16 * h + sel * 8;
        int bi = m >> 11, s0 = m & (Sn - 1);
        union { uint4 u; __bf16 hh[8]; } pk;
#pragma unroll
        for (int i = 0; i < 8; ++i) pk.hh[i] = (__bf16)(av[i] + bb);
        *reinterpret_cast<uint4*>(&vT[((size_t)bi * En + col) * Sn + s0]) = pk.u;
      } else {
        __bf16* dst = (mat == 0) ? q : k;
#pragma unroll
        for (int i = 0; i < 8; ++i) {
          int m = m0 + 16 * h + sel * 8 + i;
          dst[(size_t)m * En + col] = (__bf16)(av[i] + bb);
        }
      }
    }
  }
}

// 4) scores = q @ k^T per batch, f32 out [B][S][S]
__global__ void scores_gemm_kernel(const __bf16* __restrict__ q, const __bf16* __restrict__ k,
                                   float* __restrict__ scores) {
  const int MT = Sn / 32;   // 64
  const int NT = Sn / 64;   // 32
  int wid  = blockIdx.x * (blockDim.x >> 5) + (threadIdx.x >> 5);
  int lane = threadIdx.x & 31;
  int b   = wid / (MT * NT);
  int rem = wid % (MT * NT);
  int m0 = (rem / NT) * 32;
  int n0 = (rem % NT) * 64;

  const __bf16* A  = q + (size_t)b * Sn * En;
  const __bf16* Bp = k + (size_t)b * Sn * En;   // B(e,key)=k[key][e], contig in e
  v8f acc[8] = {};
  gemm_accum32<En, En, En>(A, Bp, m0, n0, lane, acc);

  int sel = lane >> 4, c = lane & 15;
#pragma unroll
  for (int h = 0; h < 2; ++h) {
#pragma unroll
    for (int j = 0; j < 4; ++j) {
      int col = n0 + 16 * j + c;
      v8f av = acc[4 * h + j];
#pragma unroll
      for (int i = 0; i < 8; ++i) {
        int m = m0 + 16 * h + sel * 8 + i;
        scores[((size_t)b * Sn + m) * Sn + col] = av[i];
      }
    }
  }
}

// 5) row softmax: async-stage the f32 row into LDS once, then max/sum/normalize
//    from LDS; write bf16 probs. (global_load_async_to_lds_* + s_wait_asynccnt)
__global__ void softmax_kernel(const float* __restrict__ scores, __bf16* __restrict__ probs) {
  __shared__ float srow[Sn];     // 8 KB staged row
  __shared__ float red[256];
  int t = threadIdx.x;
  const float* s = scores + (size_t)blockIdx.x * Sn;
  __bf16* p = probs + (size_t)blockIdx.x * Sn;

  // Async copy: 256 threads x 8 floats = 2048 floats (2 x b128 each).
  {
    int base = t * 8;
    unsigned lds_off = (unsigned)(size_t)&srow[base];
    const float* g = s + base;
    asm volatile(
        "global_load_async_to_lds_b128 %0, %1, off\n\t"
        "global_load_async_to_lds_b128 %0, %1, off offset:16"
        :: "v"(lds_off), "v"(g) : "memory");
    asm volatile("s_wait_asynccnt 0x0" ::: "memory");
  }
  __syncthreads();

  float lmax = -3.4e38f;
  for (int i = t; i < Sn; i += 256) lmax = fmaxf(lmax, srow[i]);
  red[t] = lmax; __syncthreads();
  for (int w = 128; w > 0; w >>= 1) { if (t < w) red[t] = fmaxf(red[t], red[t + w]); __syncthreads(); }
  float mx = red[0]; __syncthreads();

  float lsum = 0.f;
  for (int i = t; i < Sn; i += 256) lsum += __expf(srow[i] - mx);
  red[t] = lsum; __syncthreads();
  for (int w = 128; w > 0; w >>= 1) { if (t < w) red[t] += red[t + w]; __syncthreads(); }
  float inv = 1.0f / red[0];

  for (int i = t; i < Sn; i += 256) p[i] = (__bf16)(__expf(srow[i] - mx) * inv);
}

// 6) weighted = probs @ v  (v held transposed: vT[b][e][s] -> K contiguous)
__global__ void pv_gemm_kernel(const __bf16* __restrict__ probs, const __bf16* __restrict__ vT,
                               __bf16* __restrict__ wout) {
  const int MT = Sn / 32;   // 64
  const int NT = En / 64;   // 16
  int wid  = blockIdx.x * (blockDim.x >> 5) + (threadIdx.x >> 5);
  int lane = threadIdx.x & 31;
  int b   = wid / (MT * NT);
  int rem = wid % (MT * NT);
  int m0 = (rem / NT) * 32;
  int n0 = (rem % NT) * 64;

  const __bf16* A  = probs + (size_t)b * Sn * Sn;
  const __bf16* Bp = vT + (size_t)b * En * Sn;  // B(key,e)=vT[e][key], contig in key
  v8f acc[8] = {};
  gemm_accum32<Sn, Sn, Sn>(A, Bp, m0, n0, lane, acc);

  int sel = lane >> 4, c = lane & 15;
#pragma unroll
  for (int h = 0; h < 2; ++h) {
#pragma unroll
    for (int j = 0; j < 4; ++j) {
      int col = n0 + 16 * j + c;
      v8f av = acc[4 * h + j];
#pragma unroll
      for (int i = 0; i < 8; ++i) {
        int m = m0 + 16 * h + sel * 8 + i;
        wout[((size_t)b * Sn + m) * En + col] = (__bf16)av[i];
      }
    }
  }
}

// 7) out = leaky_relu(weighted @ Wf^T + bf), f32 output
__global__ void out_gemm_kernel(const __bf16* __restrict__ w, const __bf16* __restrict__ wf,
                                const float* __restrict__ bf, float* __restrict__ out) {
  const int NT = En / 64;          // 16
  int wid  = blockIdx.x * (blockDim.x >> 5) + (threadIdx.x >> 5);
  int lane = threadIdx.x & 31;
  int m0 = (wid / NT) * 32;
  int n0 = (wid % NT) * 64;

  v8f acc[8] = {};
  gemm_accum32<En, En, En>(w, wf, m0, n0, lane, acc);

  int sel = lane >> 4, c = lane & 15;
#pragma unroll
  for (int h = 0; h < 2; ++h) {
#pragma unroll
    for (int j = 0; j < 4; ++j) {
      int col = n0 + 16 * j + c;
      float bb = bf[col];
      v8f av = acc[4 * h + j];
#pragma unroll
      for (int i = 0; i < 8; ++i) {
        int m = m0 + 16 * h + sel * 8 + i;
        float val = av[i] + bb;
        out[(size_t)m * En + col] = (val >= 0.f) ? val : NEG_SLOPE * val;
      }
    }
  }
}

// ---------------------------------------------------------------------------
extern "C" void kernel_launch(void* const* d_in, const int* in_sizes, int n_in,
                              void* d_out, int out_size, void* d_ws, size_t ws_size,
                              hipStream_t stream) {
  const float* e1 = (const float*)d_in[0];
  const float* e2 = (const float*)d_in[1];
  const float* e3 = (const float*)d_in[2];
  const float* Wq = (const float*)d_in[3];
  const float* bq = (const float*)d_in[4];
  const float* Wk = (const float*)d_in[5];
  const float* bk = (const float*)d_in[6];
  const float* Wv = (const float*)d_in[7];
  const float* bv = (const float*)d_in[8];
  const float* Wf = (const float*)d_in[9];
  const float* bf = (const float*)d_in[10];
  float* out = (float*)d_out;

  // workspace carve-up (f32 scores first for alignment, then bf16 buffers)
  char* ws = (char*)d_ws;
  size_t off = 0;
  float*  scores = (float*)(ws + off);  off += (size_t)Bn * Sn * Sn * sizeof(float);   // 128MB
  __bf16* xbf    = (__bf16*)(ws + off); off += (size_t)Bn * Sn * En * 2;               // 32MB
  __bf16* wbf    = (__bf16*)(ws + off); off += (size_t)4 * En * En * 2;                // 8MB
  __bf16* qb     = (__bf16*)(ws + off); off += (size_t)Bn * Sn * En * 2;               // 32MB
  __bf16* kb     = (__bf16*)(ws + off); off += (size_t)Bn * Sn * En * 2;               // 32MB
  __bf16* vT     = (__bf16*)(ws + off); off += (size_t)Bn * Sn * En * 2;               // 32MB
  __bf16* probs  = (__bf16*)(ws + off); off += (size_t)Bn * Sn * Sn * 2;               // 64MB
  __bf16* wei    = (__bf16*)(ws + off); off += (size_t)Bn * Sn * En * 2;               // 32MB
  (void)ws_size; (void)in_sizes; (void)n_in; (void)out_size;

  const int TB = 256;                 // 8 waves per block
  const int WPB = TB / 32;

  // 1) pack x
  pack_x_kernel<<<dim3(8192), dim3(TB), 0, stream>>>(e1, e2, e3, xbf);
  // 2) pack weights
  pack_w_kernel<<<dim3(4096), dim3(TB), 0, stream>>>(Wq, Wk, Wv, Wf, wbf);
  // 3) q,k,v projections: 3 * 512 * 16 wave tiles
  {
    int waves = 3 * ((Bn * Sn) / 32) * (En / 64);
    qkv_gemm_kernel<<<dim3(waves / WPB), dim3(TB), 0, stream>>>(
        xbf, wbf, bq, bk, bv, qb, kb, vT);
  }
  // 4) scores: 8 * 64 * 32 wave tiles
  {
    int waves = Bn * (Sn / 32) * (Sn / 64);
    scores_gemm_kernel<<<dim3(waves / WPB), dim3(TB), 0, stream>>>(qb, kb, scores);
  }
  // 5) softmax over each of B*S rows
  softmax_kernel<<<dim3(Bn * Sn), dim3(TB), 0, stream>>>(scores, probs);
  // 6) probs @ v: 8 * 64 * 16 wave tiles
  {
    int waves = Bn * (Sn / 32) * (En / 64);
    pv_gemm_kernel<<<dim3(waves / WPB), dim3(TB), 0, stream>>>(probs, vT, wei);
  }
  // 7) output projection + bias + leaky relu
  {
    int waves = ((Bn * Sn) / 32) * (En / 64);
    out_gemm_kernel<<<dim3(waves / WPB), dim3(TB), 0, stream>>>(wei, wbf + (size_t)3 * En * En, bf, out);
  }
}
